// VFIformer_39668317946158
// MI455X (gfx1250) — matrix operand order repaired and checked by reading, not compiled, not run on hardware
//
#include <hip/hip_runtime.h>
#include <hip/hip_bf16.h>
#include <cstdint>
#include <cstddef>

// ---------------------------------------------------------------------------
// CDNA5 (gfx1250) implementation. Conv layers -> implicit GEMM on
// v_wmma_f32_16x16x32_f16 (wave32), 32x16 output tile per wave (2 WMMAs per
// im2col B-fragment). Weight tiles staged into LDS by the Tensor Data Mover
// (tensor_load_to_lds + s_wait_tensorcnt) when the toolchain exposes it.
// Everything else is bandwidth-bound VALU.
// ---------------------------------------------------------------------------

typedef __attribute__((ext_vector_type(16))) _Float16     v16h;
typedef __attribute__((ext_vector_type(8)))  float        v8f;
typedef __attribute__((ext_vector_type(4)))  unsigned int v4u;
typedef __attribute__((ext_vector_type(8)))  int          v8i;
typedef __attribute__((ext_vector_type(4)))  int          v4i;

#if __has_builtin(__builtin_amdgcn_tensor_load_to_lds) && \
    __has_builtin(__builtin_amdgcn_s_wait_tensorcnt)
#define VFI_HAS_TDM 1
#endif

#define KCHUNK 128
#define MTILE  32      // output channels per block (2 x 16 WMMA M-tiles)
#define ACT_NONE 0
#define ACT_PRELU 1
#define ACT_SIG 2

// ---------------------------------------------------------------------------
// Implicit-GEMM convolution: out[co][pix] = sum_k W[co][k] * im2col(x)[k][pix]
// Block = 128 threads = 4 waves. grid.x tiles pixels (64/block), grid.y tiles
// output channels (32/block). Weight K-chunks DMA'd into LDS (f32) by TDM;
// A-fragments convert f32->f16 on read. KSIZE is the compile-time conv kernel
// size (1 or 3) so im2col index decomposition is constant-divisor mul/shift.
// ---------------------------------------------------------------------------
template <int KSIZE>
__global__ __launch_bounds__(128) void vfi_conv_wmma(
    const float* __restrict__ x, const float* __restrict__ w,
    const float* __restrict__ bias, const float* __restrict__ alpha,
    float* __restrict__ out,
    int Bn, int Cin, int Hin, int Win, int Cout,
    int stride, int pad, int act)
{
  __shared__ float wLds[MTILE * KCHUNK];   // sole LDS object -> offset 0

  constexpr int KK = KSIZE * KSIZE;
  const int Ho = (Hin + 2 * pad - KSIZE) / stride + 1;
  const int Wo = (Win + 2 * pad - KSIZE) / stride + 1;
  const int Npix = Bn * Ho * Wo;
  const int Ktot = Cin * KK;
  const int KtotPad = (Ktot + 31) & ~31;

  const int lane = threadIdx.x & 31;
  const int wid  = threadIdx.x >> 5;
  const int hi   = lane >> 4;          // lane group (0: lanes 0-15, 1: 16-31)
  const int mn   = lane & 15;

  const int coBase = blockIdx.y * MTILE;
  const int pix    = (blockIdx.x * 4 + wid) * 16 + mn;   // N index for B-frag
  const bool pvalid = (pix < Npix);

  int n_ = 0, oh = 0, ow = 0;
  if (pvalid) {
    n_ = pix / (Ho * Wo);
    int r = pix - n_ * (Ho * Wo);
    oh = r / Wo; ow = r - oh * Wo;
  }
  const int ihBase = oh * stride - pad;
  const int iwBase = ow * stride - pad;
  const float* __restrict__ xb = x + (size_t)n_ * Cin * Hin * Win;

  v8f acc0 = {};
  v8f acc1 = {};

  for (int kc = 0; kc < KtotPad; kc += KCHUNK) {
    const int clen = (KCHUNK < (KtotPad - kc)) ? KCHUNK : (KtotPad - kc);

#if defined(VFI_HAS_TDM)
    // --- TDM weight staging: 2D tile (MTILE rows x clen cols, f32),
    //     row stride Ktot, OOB (k >= Ktot or row >= Cout-coBase) zero-fills.
    if (wid == 0) {
      unsigned long long ga =
          (unsigned long long)(const void*)(w + (size_t)coBase * Ktot + kc);
      const unsigned rows = (unsigned)(Cout - coBase);
      // D# group 0: count=1 | lds_addr | global_addr[56:0] | type=2
      v4u g0 = { 1u, 0u, (unsigned)ga,
                 (unsigned)(((ga >> 32) & 0x1FFFFFFull) | (2u << 30)) };
      // D# group 1: data_size=4B; tensor_dim0=Ktot; tensor_dim1=rows;
      //             tile_dim0=clen; tile_dim1=MTILE; dim0_stride=Ktot
      v8i g1 = {
        (int)(2u << 16),                                        // [31:0]
        (int)(((unsigned)Ktot & 0xFFFFu) << 16),                // [63:32]
        (int)((((unsigned)Ktot >> 16) & 0xFFFFu) |
              ((rows & 0xFFFFu) << 16)),                        // [95:64]
        (int)(((rows >> 16) & 0xFFFFu) |
              ((unsigned)clen << 16)),                          // [127:96]
        (int)MTILE,                                             // [159:128]
        (int)(unsigned)Ktot,                                    // [191:160]
        0, 0                                                    // [255:192]
      };
      v4i g2 = {};
      v4i g3 = {};
#if __clang_major__ >= 23
      v8i g4 = {};
      __builtin_amdgcn_tensor_load_to_lds(g0, g1, g2, g3, g4, 0);
#else
      __builtin_amdgcn_tensor_load_to_lds(g0, g1, g2, g3, 0);
#endif
      __builtin_amdgcn_s_wait_tensorcnt(0);
    }
#else
    // --- fallback: cooperative weight staging into LDS ---------------------
    for (int t = threadIdx.x; t < MTILE * clen; t += 128) {
      int mm = t / clen;
      int kk = t - mm * clen;
      int kg = kc + kk;
      int co = coBase + mm;
      float v = 0.f;
      if (kg < Ktot && co < Cout) v = w[(size_t)co * Ktot + kg];
      wLds[mm * clen + kk] = v;
    }
    if (kc + KCHUNK < Ktot) {
      const float* nxt = w + (size_t)coBase * Ktot + kc + KCHUNK
                       + (threadIdx.x & 127) * 16;
      __builtin_prefetch(nxt, 0, 1);
    }
#endif
    __syncthreads();

    #pragma unroll 2
    for (int ks = 0; ks < clen; ks += 32) {
      // --- B fragment: 32x16 f16 im2col gather from global -----------------
      // lane holds column N=pix; element e -> K = e + 16*hi
      v16h bf = {};
      #pragma unroll
      for (int e = 0; e < 16; ++e) {
        int kg = kc + ks + e + hi * 16;
        float v = 0.f;
        if (pvalid && (kg < Ktot)) {
          int ci, ih, iw;
          if (KSIZE == 1) {
            ci = kg; ih = ihBase; iw = iwBase;
          } else {
            ci = kg / KK;                 // constant divisor -> mul/shift
            int r2 = kg - ci * KK;
            int kh = r2 / KSIZE, kw = r2 - kh * KSIZE;
            ih = ihBase + kh; iw = iwBase + kw;
          }
          if ((unsigned)ih < (unsigned)Hin && (unsigned)iw < (unsigned)Win)
            v = xb[((size_t)ci * Hin + ih) * Win + iw];
        }
        bf[e] = (_Float16)v;
      }
      // --- A fragments: two 16x32 weight tiles from LDS (f32 -> f16) -------
      // lane m holds row M=m; element e -> K = e + 16*hi
      v16h af0 = {}, af1 = {};
      #pragma unroll
      for (int e = 0; e < 16; e += 2) {
        int kloc = e + hi * 16;
        af0[e]     = (_Float16)wLds[mn * clen + ks + kloc];
        af0[e + 1] = (_Float16)wLds[mn * clen + ks + kloc + 1];
        af1[e]     = (_Float16)wLds[(16 + mn) * clen + ks + kloc];
        af1[e + 1] = (_Float16)wLds[(16 + mn) * clen + ks + kloc + 1];
      }
      acc0 = __builtin_amdgcn_wmma_f32_16x16x32_f16(
          false, af0, false, bf, (short)0, acc0, false, false);
      acc1 = __builtin_amdgcn_wmma_f32_16x16x32_f16(
          false, af1, false, bf, (short)0, acc1, false, false);
    }
    __syncthreads();
  }

  // --- epilogue: bias + activation, D layout: lane n=N, vgpr v -> M=v+8*hi --
  if (pvalid) {
    float* __restrict__ ob = out + ((size_t)n_ * Cout) * Ho * Wo
                           + (size_t)oh * Wo + ow;
    const size_t cs = (size_t)Ho * Wo;
    #pragma unroll
    for (int v = 0; v < 8; ++v) {
      #pragma unroll
      for (int mtile = 0; mtile < 2; ++mtile) {
        int co = coBase + mtile * 16 + v + 8 * hi;
        if (co < Cout) {
          float r = (mtile == 0 ? acc0[v] : acc1[v]) + bias[co];
          if (act == ACT_PRELU) {
            float al = alpha[co];
            r = (r >= 0.f) ? r : al * r;
          } else if (act == ACT_SIG) {
            r = 1.f / (1.f + __expf(-r));
          }
          ob[(size_t)co * cs] = r;
        }
      }
    }
  }
}

// ---------------------------------------------------------------------------
// Bilinear backward warp (clamped coords). flow layout: (B, flowC, H, W),
// channels 0/1 of the passed pointer are (dx, dy).
// ---------------------------------------------------------------------------
__global__ void vfi_warp(const float* __restrict__ x,
                         const float* __restrict__ flow, int flowC,
                         float* __restrict__ out, int B, int C, int H, int W)
{
  int i = blockIdx.x * blockDim.x + threadIdx.x;
  int total = B * C * H * W;
  if (i >= total) return;
  int w_ = i % W; int t = i / W;
  int h_ = t % H; t /= H;
  int c  = t % C; int b = t / C;
  size_t fb = ((size_t)b * flowC) * H * W + (size_t)h_ * W + w_;
  float fx = flow[fb];
  float fy = flow[fb + (size_t)H * W];
  float px = fminf(fmaxf((float)w_ + fx, 0.f), (float)(W - 1));
  float py = fminf(fmaxf((float)h_ + fy, 0.f), (float)(H - 1));
  float x0f = floorf(px), y0f = floorf(py);
  float wx = px - x0f, wy = py - y0f;
  int x0 = (int)x0f, y0 = (int)y0f;
  int x1 = min(x0 + 1, W - 1), y1 = min(y0 + 1, H - 1);
  const float* img = x + ((size_t)b * C + c) * H * W;
  float v00 = img[(size_t)y0 * W + x0];
  float v01 = img[(size_t)y0 * W + x1];
  float v10 = img[(size_t)y1 * W + x0];
  float v11 = img[(size_t)y1 * W + x1];
  out[i] = v00 * (1.f - wx) * (1.f - wy) + v01 * wx * (1.f - wy)
         + v10 * (1.f - wx) * wy + v11 * wx * wy;
}

// ---------------------------------------------------------------------------
// Normalized 9-tap correlation, taps computed on the fly (avoids materializing
// the 9x unfold tensor). out: (B, 9, H, W).
// corr[p] = sum_c fn[c] * tap[c] / sqrt(sum_c tap[c]^2 + 1e-12)
// tap = bilinear sample (zero OOB) of x shifted by (di,dj).
// ---------------------------------------------------------------------------
__global__ void vfi_corr(const float* __restrict__ x,
                         const float* __restrict__ flow, int flowC,
                         const float* __restrict__ fn,
                         float* __restrict__ out, int B, int C, int H, int W)
{
  int i = blockIdx.x * blockDim.x + threadIdx.x;
  int total = B * 9 * H * W;
  if (i >= total) return;
  int w_ = i % W; int t = i / W;
  int h_ = t % H; t /= H;
  int p  = t % 9; int b = t / 9;
  int di = p / 3 - 1, dj = p % 3 - 1;

  size_t fb = ((size_t)b * flowC) * H * W + (size_t)h_ * W + w_;
  float fx = flow[fb];
  float fy = flow[fb + (size_t)H * W];
  float px = fminf(fmaxf((float)w_ + fx, 0.f), (float)(W - 1));
  float py = fminf(fmaxf((float)h_ + fy, 0.f), (float)(H - 1));
  float x0f = floorf(px), y0f = floorf(py);
  float wx = px - x0f, wy = py - y0f;
  int x0 = (int)x0f, y0 = (int)y0f;
  int x1 = min(x0 + 1, W - 1), y1 = min(y0 + 1, H - 1);
  // shifted-image integer coords -> source coords
  int sy0 = y0 + di, sy1 = y1 + di;
  int sx0 = x0 + dj, sx1 = x1 + dj;
  float w00 = (1.f - wx) * (1.f - wy), w01 = wx * (1.f - wy);
  float w10 = (1.f - wx) * wy,        w11 = wx * wy;
  bool i00 = (unsigned)sy0 < (unsigned)H && (unsigned)sx0 < (unsigned)W;
  bool i01 = (unsigned)sy0 < (unsigned)H && (unsigned)sx1 < (unsigned)W;
  bool i10 = (unsigned)sy1 < (unsigned)H && (unsigned)sx0 < (unsigned)W;
  bool i11 = (unsigned)sy1 < (unsigned)H && (unsigned)sx1 < (unsigned)W;

  float dot = 0.f, ss = 0.f;
  size_t hw = (size_t)H * W;
  size_t pxy = (size_t)h_ * W + w_;
  const float* imgb = x + (size_t)b * C * hw;
  const float* fnb  = fn + (size_t)b * C * hw;
  for (int c = 0; c < C; ++c) {
    const float* img = imgb + (size_t)c * hw;
    float v00 = i00 ? img[(size_t)sy0 * W + sx0] : 0.f;
    float v01 = i01 ? img[(size_t)sy0 * W + sx1] : 0.f;
    float v10 = i10 ? img[(size_t)sy1 * W + sx0] : 0.f;
    float v11 = i11 ? img[(size_t)sy1 * W + sx1] : 0.f;
    float val = v00 * w00 + v01 * w01 + v10 * w10 + v11 * w11;
    dot += fnb[(size_t)c * hw + pxy] * val;
    ss  += val * val;
  }
  out[i] = dot * rsqrtf(ss + 1e-12f);
}

// Bilinear resize (half-pixel centers) with post scale.
__global__ void vfi_resize(const float* __restrict__ src, float* __restrict__ dst,
                           int B, int C, int Hs, int Ws, int Hd, int Wd, float scale)
{
  int i = blockIdx.x * blockDim.x + threadIdx.x;
  int total = B * C * Hd * Wd;
  if (i >= total) return;
  int w_ = i % Wd; int t = i / Wd;
  int h_ = t % Hd; t /= Hd;
  int c  = t % C;  int b = t / C;
  float sx = ((float)w_ + 0.5f) * (float)Ws / (float)Wd - 0.5f;
  float sy = ((float)h_ + 0.5f) * (float)Hs / (float)Hd - 0.5f;
  sx = fminf(fmaxf(sx, 0.f), (float)(Ws - 1));
  sy = fminf(fmaxf(sy, 0.f), (float)(Hs - 1));
  float x0f = floorf(sx), y0f = floorf(sy);
  float wx = sx - x0f, wy = sy - y0f;
  int x0 = (int)x0f, y0 = (int)y0f;
  int x1 = min(x0 + 1, Ws - 1), y1 = min(y0 + 1, Hs - 1);
  const float* img = src + ((size_t)b * C + c) * Hs * Ws;
  float v = img[(size_t)y0 * Ws + x0] * (1.f - wx) * (1.f - wy)
          + img[(size_t)y0 * Ws + x1] * wx * (1.f - wy)
          + img[(size_t)y1 * Ws + x0] * (1.f - wx) * wy
          + img[(size_t)y1 * Ws + x1] * wx * wy;
  dst[i] = v * scale;
}

// fea = fea0*occl + fea1*(1-occl); occl is 1-channel broadcast
__global__ void vfi_blend(float* __restrict__ out, const float* __restrict__ f0,
                          const float* __restrict__ f1, const float* __restrict__ occl,
                          int B, int C, int HW)
{
  int i = blockIdx.x * blockDim.x + threadIdx.x;
  int total = B * C * HW;
  if (i >= total) return;
  int p = i % HW; int t = i / HW;
  int b = t / C;
  float o = occl[(size_t)b * HW + p];
  out[i] = f0[i] * o + f1[i] * (1.f - o);
}

// fn = fea / sqrt(sum_c fea^2 + 1e-12)
__global__ void vfi_l2norm(float* __restrict__ out, const float* __restrict__ in,
                           int B, int C, int HW)
{
  int i = blockIdx.x * blockDim.x + threadIdx.x;
  int total = B * HW;
  if (i >= total) return;
  int p = i % HW; int b = i / HW;
  const float* src = in + (size_t)b * C * HW + p;
  float ss = 0.f;
  for (int c = 0; c < C; ++c) { float v = src[(size_t)c * HW]; ss += v * v; }
  float inv = rsqrtf(ss + 1e-12f);
  float* dst = out + (size_t)b * C * HW + p;
  for (int c = 0; c < C; ++c) dst[(size_t)c * HW] = src[(size_t)c * HW] * inv;
}

// dst(B,n,HW) = src(B,Ct,HW)[:, off:off+n]
__global__ void vfi_slice(float* __restrict__ dst, const float* __restrict__ src,
                          int B, int Ct, int off, int n, int HW)
{
  int i = blockIdx.x * blockDim.x + threadIdx.x;
  int total = B * n * HW;
  if (i >= total) return;
  int p = i % HW; int t = i / HW;
  int c = t % n;  int b = t / n;
  dst[i] = src[((size_t)b * Ct + off + c) * HW + p];
}

// channel-concat of up to 4 sources
__global__ void vfi_concat4(float* __restrict__ dst,
                            const float* __restrict__ s0, int c0,
                            const float* __restrict__ s1, int c1,
                            const float* __restrict__ s2, int c2,
                            const float* __restrict__ s3, int c3,
                            int B, int HW)
{
  int Ct = c0 + c1 + c2 + c3;
  int i = blockIdx.x * blockDim.x + threadIdx.x;
  int total = B * Ct * HW;
  if (i >= total) return;
  int p = i % HW; int t = i / HW;
  int c = t % Ct; int b = t / Ct;
  float v;
  if (c < c0)                v = s0[((size_t)b * c0 + c) * HW + p];
  else if (c < c0 + c1)      v = s1[((size_t)b * c1 + (c - c0)) * HW + p];
  else if (c < c0 + c1 + c2) v = s2[((size_t)b * c2 + (c - c0 - c1)) * HW + p];
  else                       v = s3[((size_t)b * c3 + (c - c0 - c1 - c2)) * HW + p];
  dst[i] = v;
}

__global__ void vfi_add(float* __restrict__ y, const float* __restrict__ x, int n)
{
  int i = blockIdx.x * blockDim.x + threadIdx.x;
  if (i < n) y[i] += x[i];
}

// ---------------------------------------------------------------------------
// Host orchestration
// ---------------------------------------------------------------------------
namespace {
struct CPp { const float *a, *b, *w; };

// sorted-key offsets within an rf param block (each entry = 3 tensors a,b,w)
enum { O_CORR1 = 0, O_CORR2 = 3, O_FLOWC1 = 6, O_FLOWC2 = 9, O_GRU1 = 12,
       O_GRU2 = 15, O_HEAD1 = 18, O_HEAD2 = 21, O_MOTION = 24, O_OCCL1 = 27,
       O_OCCL2 = 30, O_OCCL3 = 33 };
}

extern "C" void kernel_launch(void* const* d_in, const int* in_sizes, int n_in,
                              void* d_out, int out_size, void* d_ws, size_t ws_size,
                              hipStream_t stream)
{
  (void)in_sizes; (void)n_in; (void)out_size; (void)ws_size;
  const int B = 4, H = 192, W = 192, C0 = 24;

  auto getCP = [&](int base) -> CPp {
    return { (const float*)d_in[base], (const float*)d_in[base + 1],
             (const float*)d_in[base + 2] };
  };
  const float* x0in   = (const float*)d_in[0];
  const float* x1in   = (const float*)d_in[1];
  const float* flowin = (const float*)d_in[2];
  // encoder params (sorted keys): c1a..c4b starting at index 3
  CPp enc[8];
  for (int e = 0; e < 8; ++e) enc[e] = getCP(3 + 3 * e);
  const int rfBase[4] = { 135, 99, 63, 27 };   // rf4, rf3, rf2, rf1

  char* ws = (char*)d_ws;
  size_t off = 0;
  auto alloc = [&](size_t elems) -> float* {
    float* p = (float*)(ws + off);
    off = (off + elems * sizeof(float) + 255) & ~(size_t)255;
    return p;
  };

  auto conv = [&](const float* x, CPp p, float* out, int Bn, int Cin,
                  int Hin, int Win, int Cout, int k, int stride, int act) {
    int pad = (k - 1) / 2;
    int Ho = (Hin + 2 * pad - k) / stride + 1;
    int Wo = (Win + 2 * pad - k) / stride + 1;
    int Np = Bn * Ho * Wo;
    dim3 g((Np + 63) / 64, (Cout + MTILE - 1) / MTILE);
    if (k == 1)
      vfi_conv_wmma<1><<<g, 128, 0, stream>>>(x, p.w, p.b, p.a, out, Bn, Cin,
                                              Hin, Win, Cout, stride, pad, act);
    else
      vfi_conv_wmma<3><<<g, 128, 0, stream>>>(x, p.w, p.b, p.a, out, Bn, Cin,
                                              Hin, Win, Cout, stride, pad, act);
  };
  auto launch1d = [&](int total) { return dim3((total + 255) / 256); };

  // ---------------- persistent buffers ------------------------------------
  float* s1 = alloc((size_t)8 * C0 * H * W);
  float* s2 = alloc((size_t)8 * 2 * C0 * (H/2) * (W/2));
  float* s3 = alloc((size_t)8 * 4 * C0 * (H/4) * (W/4));
  float* s4 = alloc((size_t)8 * 8 * C0 * (H/8) * (W/8));
  float* fA = alloc((size_t)B * 4 * H * W);
  float* fB = alloc((size_t)B * 4 * H * W);
  size_t mark = off;

  // ---------------- encoder ------------------------------------------------
  {
    float* inp = alloc((size_t)8 * 3 * H * W);
    float* tA  = alloc((size_t)8 * C0 * H * W);
    size_t half = (size_t)B * 3 * H * W * sizeof(float);
    hipMemcpyAsync(inp, x0in, half, hipMemcpyDeviceToDevice, stream);
    hipMemcpyAsync((char*)inp + half, x1in, half, hipMemcpyDeviceToDevice, stream);
    conv(inp, enc[0], tA, 8, 3,    H,   W,   C0,   3, 1, ACT_PRELU);
    conv(tA,  enc[1], s1, 8, C0,   H,   W,   C0,   3, 1, ACT_PRELU);
    conv(s1,  enc[2], tA, 8, C0,   H,   W,   2*C0, 3, 2, ACT_PRELU);
    conv(tA,  enc[3], s2, 8, 2*C0, H/2, W/2, 2*C0, 3, 1, ACT_PRELU);
    conv(s2,  enc[4], tA, 8, 2*C0, H/2, W/2, 4*C0, 3, 2, ACT_PRELU);
    conv(tA,  enc[5], s3, 8, 4*C0, H/4, W/4, 4*C0, 3, 1, ACT_PRELU);
    conv(s3,  enc[6], tA, 8, 4*C0, H/4, W/4, 8*C0, 3, 2, ACT_PRELU);
    conv(tA,  enc[7], s4, 8, 8*C0, H/8, W/8, 8*C0, 3, 1, ACT_PRELU);
  }
  off = mark;

  // ---------------- initial flow at s4 resolution --------------------------
  float* fcur = fA;
  float* foth = fB;
  vfi_resize<<<launch1d(B * 4 * (H/8) * (W/8)), 256, 0, stream>>>(
      flowin, fcur, B, 4, H/2, W/2, H/8, W/8, 0.25f);

  const int   Cs[4] = { 8*C0, 4*C0, 2*C0, C0 };
  const int   Hl[4] = { H/8, H/4, H/2, H };
  float* const sv[4] = { s4, s3, s2, s1 };

  for (int li = 0; li < 4; ++li) {
    const int C = Cs[li], Hc = Hl[li], Wc = Hl[li];
    const int HW = Hc * Wc;
    if (li > 0) {
      vfi_resize<<<launch1d(B * 4 * HW), 256, 0, stream>>>(
          fcur, foth, B, 4, Hl[li-1], Hl[li-1], Hc, Wc, 2.0f);
      float* t = fcur; fcur = foth; foth = t;
    }
    const int rb = rfBase[li];
    CPp Pcorr1 = getCP(rb + O_CORR1),  Pcorr2 = getCP(rb + O_CORR2);
    CPp Pfc1   = getCP(rb + O_FLOWC1), Pfc2   = getCP(rb + O_FLOWC2);
    CPp Pg1    = getCP(rb + O_GRU1),   Pg2    = getCP(rb + O_GRU2);
    CPp Ph1    = getCP(rb + O_HEAD1),  Ph2    = getCP(rb + O_HEAD2);
    CPp Pmot   = getCP(rb + O_MOTION);
    CPp Po1    = getCP(rb + O_OCCL1),  Po2 = getCP(rb + O_OCCL2);
    CPp Po3    = getCP(rb + O_OCCL3);

    const float* x0lev = sv[li];
    const float* x1lev = sv[li] + (size_t)B * C * HW;

    off = mark;
    float* f0   = alloc((size_t)B * 2 * HW);
    float* f1   = alloc((size_t)B * 2 * HW);
    float* fea0 = alloc((size_t)B * C * HW);
    float* fea1 = alloc((size_t)B * C * HW);
    float* fea  = alloc((size_t)B * C * HW);
    float* fn   = alloc((size_t)B * C * HW);
    float* ta   = alloc((size_t)B * C * HW);
    float* tb   = alloc((size_t)B * C * HW);
    float* c0f  = alloc((size_t)B * C * HW);
    float* c1f  = alloc((size_t)B * C * HW);
    float* ff0  = alloc((size_t)B * C * HW);
    float* ff1  = alloc((size_t)B * C * HW);
    float* m0   = alloc((size_t)B * C * HW);
    float* m1   = alloc((size_t)B * C * HW);
    float* cat2 = alloc((size_t)B * 2 * C * HW);
    float* occl = alloc((size_t)B * HW);
    float* cr0  = alloc((size_t)B * 9 * HW);
    float* cr1  = alloc((size_t)B * 9 * HW);
    float* gin  = alloc((size_t)B * (3 * C + 2) * HW);
    float* dlt  = alloc((size_t)B * 2 * HW);

    vfi_slice<<<launch1d(B*2*HW), 256, 0, stream>>>(f0, fcur, B, 4, 0, 2, HW);
    vfi_slice<<<launch1d(B*2*HW), 256, 0, stream>>>(f1, fcur, B, 4, 2, 2, HW);

    vfi_warp<<<launch1d(B*C*HW), 256, 0, stream>>>(x0lev, f0, 2, fea0, B, C, Hc, Wc);
    vfi_warp<<<launch1d(B*C*HW), 256, 0, stream>>>(x1lev, f1, 2, fea1, B, C, Hc, Wc);

    vfi_concat4<<<launch1d(B*2*C*HW), 256, 0, stream>>>(
        cat2, fea0, C, fea1, C, nullptr, 0, nullptr, 0, B, HW);
    conv(cat2, Po1, ta, B, 2*C, Hc, Wc, C, 1, 1, ACT_PRELU);
    conv(ta,   Po2, tb, B, C,   Hc, Wc, C, 1, 1, ACT_PRELU);
    conv(tb,   Po3, occl, B, C, Hc, Wc, 1, 1, 1, ACT_SIG);

    vfi_blend<<<launch1d(B*C*HW), 256, 0, stream>>>(fea, fea0, fea1, occl, B, C, HW);
    vfi_l2norm<<<launch1d(B*HW), 256, 0, stream>>>(fn, fea, B, C, HW);

    vfi_corr<<<launch1d(B*9*HW), 256, 0, stream>>>(x0lev, f0, 2, fn, cr0, B, C, Hc, Wc);
    vfi_corr<<<launch1d(B*9*HW), 256, 0, stream>>>(x1lev, f1, 2, fn, cr1, B, C, Hc, Wc);

    conv(cr0, Pcorr1, ta,  B, 9, Hc, Wc, C, 1, 1, ACT_PRELU);
    conv(ta,  Pcorr2, c0f, B, C, Hc, Wc, C, 1, 1, ACT_PRELU);
    conv(cr1, Pcorr1, ta,  B, 9, Hc, Wc, C, 1, 1, ACT_PRELU);
    conv(ta,  Pcorr2, c1f, B, C, Hc, Wc, C, 1, 1, ACT_PRELU);

    conv(f0, Pfc1, ta,  B, 2, Hc, Wc, C, 3, 1, ACT_PRELU);
    conv(ta, Pfc2, ff0, B, C, Hc, Wc, C, 3, 1, ACT_PRELU);
    conv(f1, Pfc1, ta,  B, 2, Hc, Wc, C, 3, 1, ACT_PRELU);
    conv(ta, Pfc2, ff1, B, C, Hc, Wc, C, 3, 1, ACT_PRELU);

    vfi_concat4<<<launch1d(B*2*C*HW), 256, 0, stream>>>(
        cat2, c0f, C, ff0, C, nullptr, 0, nullptr, 0, B, HW);
    conv(cat2, Pmot, m0, B, 2*C, Hc, Wc, C, 3, 1, ACT_PRELU);
    vfi_concat4<<<launch1d(B*2*C*HW), 256, 0, stream>>>(
        cat2, c1f, C, ff1, C, nullptr, 0, nullptr, 0, B, HW);
    conv(cat2, Pmot, m1, B, 2*C, Hc, Wc, C, 3, 1, ACT_PRELU);

    // head 0
    vfi_concat4<<<launch1d(B*(3*C+2)*HW), 256, 0, stream>>>(
        gin, fea, C, fea0, C, m0, C, f0, 2, B, HW);
    conv(gin, Pg1, ta,  B, 3*C+2, Hc, Wc, C, 3, 1, ACT_PRELU);
    conv(ta,  Pg2, tb,  B, C,     Hc, Wc, C, 3, 1, ACT_PRELU);
    conv(tb,  Ph1, ta,  B, C,     Hc, Wc, C, 3, 1, ACT_PRELU);
    conv(ta,  Ph2, dlt, B, C,     Hc, Wc, 2, 3, 1, ACT_NONE);
    vfi_add<<<launch1d(B*2*HW), 256, 0, stream>>>(f0, dlt, B * 2 * HW);
    // head 1
    vfi_concat4<<<launch1d(B*(3*C+2)*HW), 256, 0, stream>>>(
        gin, fea, C, fea1, C, m1, C, f1, 2, B, HW);
    conv(gin, Pg1, ta,  B, 3*C+2, Hc, Wc, C, 3, 1, ACT_PRELU);
    conv(ta,  Pg2, tb,  B, C,     Hc, Wc, C, 3, 1, ACT_PRELU);
    conv(tb,  Ph1, ta,  B, C,     Hc, Wc, C, 3, 1, ACT_PRELU);
    conv(ta,  Ph2, dlt, B, C,     Hc, Wc, 2, 3, 1, ACT_NONE);
    vfi_add<<<launch1d(B*2*HW), 256, 0, stream>>>(f1, dlt, B * 2 * HW);

    vfi_concat4<<<launch1d(B*4*HW), 256, 0, stream>>>(
        foth, f0, 2, f1, 2, nullptr, 0, nullptr, 0, B, HW);
    float* t = fcur; fcur = foth; foth = t;
  }

  // ---------------- outputs ------------------------------------------------
  float* outp = (float*)d_out;
  size_t ooff = 0;
  hipMemcpyAsync(outp, fcur, (size_t)B * 4 * H * W * sizeof(float),
                 hipMemcpyDeviceToDevice, stream);
  ooff += (size_t)B * 4 * H * W;

  off = mark;
  float* flA = alloc((size_t)B * 2 * H * W);
  float* flB = alloc((size_t)B * 2 * H * W);

  for (int side = 0; side < 2; ++side) {
    int HW0 = H * W;
    vfi_slice<<<launch1d(B*2*HW0), 256, 0, stream>>>(
        flA, fcur, B, 4, side * 2, 2, HW0);
    float* fl = flA; float* flo = flB;
    for (int lv = 0; lv < 4; ++lv) {
      int C = Cs[3 - lv];                 // s1..s4 channels: 24,48,96,192
      int Hc = Hl[3 - lv], Wc = Hl[3 - lv];
      int HW = Hc * Wc;
      const float* feas = sv[3 - lv] + (size_t)side * B * C * HW;
      vfi_warp<<<launch1d(B*C*HW), 256, 0, stream>>>(
          feas, fl, 2, outp + ooff, B, C, Hc, Wc);
      ooff += (size_t)B * C * HW;
      if (lv < 3) {
        vfi_resize<<<launch1d(B*2*(HW/4)), 256, 0, stream>>>(
            fl, flo, B, 2, Hc, Wc, Hc / 2, Wc / 2, 0.5f);
        float* t = fl; fl = flo; flo = t;
      }
    }
  }
}
